// SSIMLoss_27350351741297
// MI455X (gfx1250) — matrix operand order repaired and checked by reading, not compiled, not run on hardware
//
#include <hip/hip_runtime.h>
#include <hip/hip_fp16.h>

typedef __attribute__((ext_vector_type(16))) _Float16 v16h;
typedef __attribute__((ext_vector_type(8)))  float    v8f;

#define IMG_H 512
#define IMG_W 512
#define NPLANES 48            // B*C = 16*3
#define TILES_PER_PLANE 1024  // 32*32 tiles of 16x16
#define PAD 5
#define WAVES_PER_BLOCK 8

// normalized 11-tap gaussian, sigma=1.5:  g(d) = exp(-(d-5)^2/4.5)/sum, d in [0,10]
__device__ __forceinline__ float gauss_w(int d) {
  if (d < 0 || d > 10) return 0.0f;
  float x = (float)(d - 5);
  return __expf(x * x * (-1.0f / 4.5f)) * 0.2660117227f; // 1/3.7592330
}

__device__ __forceinline__ float ld_pix(const float* __restrict__ p, int r, int c) {
  return ((unsigned)r < (unsigned)IMG_H && (unsigned)c < (unsigned)IMG_W)
             ? p[r * IMG_W + c] : 0.0f;
}

// Separable 11-tap conv of one 16x16 tile:
//   Y = A_img(16x32 rows x K) @ Gfrag(32x16)  (two row-halves)
//   Z = Afrag(16x32)          @ Y(32x16)
// a[0] = region rows 0..15, a[1] = region rows 16..31 (rows >=26 zeroed).
__device__ __forceinline__ v8f conv_tile(const v16h* a, v16h Gfrag, v16h Afrag, int lane) {
  v8f zc = {};
  v8f ylo = __builtin_amdgcn_wmma_f32_16x16x32_f16(false, a[0], false, Gfrag,
                                                   (short)0, zc, false, false);
  v8f yhi = __builtin_amdgcn_wmma_f32_16x16x32_f16(false, a[1], false, Gfrag,
                                                   (short)0, zc, false, false);
  // C-layout (lane n: col n%16, VGPR j: row j + 8*(n>=16)) -> B-layout
  // (lane n<16: K=0..15 of col n; lane n>=16: K=16..31 of col n-16)
  const bool hi16 = lane >= 16;
  v16h b;
#pragma unroll
  for (int j = 0; j < 8; ++j) {
    float lo = ylo[j];
    float hv = yhi[j];
    float plo = __shfl_xor(lo, 16, 32); // partner's rows 8..15   (for lanes <16)
    float phi = __shfl_xor(hv, 16, 32); // partner's rows 16..23 (for lanes >=16)
    b[j]     = (_Float16)(hi16 ? phi : lo);  // K = j     | K = 16+j
    b[j + 8] = (_Float16)(hi16 ? hv  : plo); // K = j+8   | K = 24+j
  }
  return __builtin_amdgcn_wmma_f32_16x16x32_f16(false, Afrag, false, b,
                                                (short)0, zc, false, false);
}

__global__ void __launch_bounds__(256)
ssim_tile_kernel(const float* __restrict__ img1, const float* __restrict__ img2,
                 double* __restrict__ acc) {
  const int lane = threadIdx.x & 31;
  const int tile = blockIdx.x * WAVES_PER_BLOCK + (threadIdx.x >> 5);
  const int plane = tile >> 10;
  const int tIn   = tile & (TILES_PER_PLANE - 1);
  const int ty = tIn >> 5, tx = tIn & 31;
  const float* __restrict__ p1 = img1 + (size_t)plane * IMG_H * IMG_W;
  const float* __restrict__ p2 = img2 + (size_t)plane * IMG_H * IMG_W;
  const int row0 = ty * 16 - PAD;
  const int col0 = tx * 16 - PAD;

  const int  mrow = lane & 15;
  const bool hi16 = lane >= 16;

  // Constant band matrices (built per-lane; Gfrag K in AMD table order, Afrag K contiguous to match b):
  //  Gfrag: B-layout 32x16, G[k][n] = g(k-n)   (horizontal taps)
  //  Afrag: A-layout 16x32, A[m][i] = g(i-m)   (vertical taps)
  v16h Gfrag, Afrag;
#pragma unroll
  for (int j = 0; j < 16; ++j) {
    int kB = ((j < 8) ? j : (j + 8)) + (hi16 ? 8 : 0); // B-layout K index
    Gfrag[j] = (_Float16)gauss_w(kB - mrow);
    int kA = hi16 ? (16 + j) : j;                    // A-layout K index
    Afrag[j] = (_Float16)gauss_w(kA - mrow);
  }

  // Load 26x26 halo region (zero-padded) as two 16x32 A-fragments per quantity.
  v16h a1[2], a2[2], a11[2], a22[2], a12[2];
#pragma unroll
  for (int grp = 0; grp < 2; ++grp) {
    const int  rr = row0 + mrow + 16 * grp;
    const bool rowLive = (grp == 0) || (mrow < 10); // region rows 26..31 unused
#pragma unroll
    for (int j = 0; j < 16; ++j) {
      const int k = ((j < 8) ? j : (j + 8)) + (hi16 ? 8 : 0);
      float x1 = 0.0f, x2 = 0.0f;
      if (rowLive && k <= 25) { // region cols 26..31 hit zero band weights
        x1 = ld_pix(p1, rr, col0 + k);
        x2 = ld_pix(p2, rr, col0 + k);
      }
      a1[grp][j]  = (_Float16)x1;
      a2[grp][j]  = (_Float16)x2;
      a11[grp][j] = (_Float16)(x1 * x1);
      a22[grp][j] = (_Float16)(x2 * x2);
      a12[grp][j] = (_Float16)(x1 * x2);
    }
  }

  // Five separable convolutions (15 WMMAs total)
  v8f mu1 = conv_tile(a1,  Gfrag, Afrag, lane);
  v8f mu2 = conv_tile(a2,  Gfrag, Afrag, lane);
  v8f e11 = conv_tile(a11, Gfrag, Afrag, lane);
  v8f e22 = conv_tile(a22, Gfrag, Afrag, lane);
  v8f e12 = conv_tile(a12, Gfrag, Afrag, lane);

  const float C1 = 1.0e-4f;  // (0.01*range)^2
  const float C2 = 9.0e-4f;  // (0.03*range)^2
  float s = 0.0f;
#pragma unroll
  for (int j = 0; j < 8; ++j) {
    float m1 = mu1[j], m2 = mu2[j];
    float m1s = m1 * m1, m2s = m2 * m2, m12 = m1 * m2;
    float s1  = e11[j] - m1s;
    float s2  = e22[j] - m2s;
    float s12 = e12[j] - m12;
    float num = (2.0f * m12 + C1) * (2.0f * s12 + C2);
    float den = (m1s + m2s + C1) * (s1 + s2 + C2);
    s += num / den;
  }

  // wave32 reduction, one f64 atomic per tile
#pragma unroll
  for (int off = 16; off >= 1; off >>= 1) s += __shfl_xor(s, off, 32);
  if (lane == 0) atomicAdd(acc, (double)s);
}

__global__ void ssim_init(double* acc) { acc[0] = 0.0; }

__global__ void ssim_finalize(const double* __restrict__ acc, float* __restrict__ out) {
  const double n = (double)NPLANES * IMG_H * IMG_W; // 12,582,912
  out[0] = (float)(1.0 - acc[0] / n);
}

extern "C" void kernel_launch(void* const* d_in, const int* in_sizes, int n_in,
                              void* d_out, int out_size, void* d_ws, size_t ws_size,
                              hipStream_t stream) {
  const float* img1 = (const float*)d_in[0];
  const float* img2 = (const float*)d_in[1];
  float*  out = (float*)d_out;
  double* acc = (double*)d_ws;

  const int total_tiles = NPLANES * TILES_PER_PLANE;          // 49152
  const int blocks = total_tiles / WAVES_PER_BLOCK;           // 6144

  ssim_init<<<1, 1, 0, stream>>>(acc);
  ssim_tile_kernel<<<blocks, 256, 0, stream>>>(img1, img2, acc);
  ssim_finalize<<<1, 1, 0, stream>>>(acc, out);
}
